// Block_40535901340278
// MI455X (gfx1250) — matrix operand order repaired and checked by reading, not compiled, not run on hardware
//
#include <hip/hip_runtime.h>

// ---------------------------------------------------------------------------
// Types for CDNA5 WMMA (wave32): v_wmma_f32_16x16x32_bf16
// ---------------------------------------------------------------------------
typedef __attribute__((ext_vector_type(16))) __bf16 bf16x16;
typedef __attribute__((ext_vector_type(8)))  __bf16 bf16x8;
typedef __attribute__((ext_vector_type(8)))  float  f32x8;

#define B_ 8
#define S_ 1024
#define E_ 1024
#define H_ 16
#define D_ 64
#define F_ 4096

// fp32 -> bf16 bits, round-to-nearest-even
static __device__ __forceinline__ unsigned short f2bf(float f) {
  union { float f; unsigned int u; } c;
  c.f = f;
  unsigned int u = c.u;
  u += 0x7FFFu + ((u >> 16) & 1u);
  return (unsigned short)(u >> 16);
}

static __device__ __forceinline__ f32x8 wmma_bf16(bf16x16 a, bf16x16 b, f32x8 c) {
  // D = A(16x32 bf16) * B(32x16 bf16) + C(16x16 f32)
  return __builtin_amdgcn_wmma_f32_16x16x32_bf16(
      /*neg_a=*/false, a, /*neg_b=*/false, b,
      /*c_mod=*/(short)0, c, /*reuse_a=*/false, /*reuse_b=*/false);
}

// A fragment (16x32, MxK), row-major source with leading dim ld (elements).
// lanes 0-15: row M=lane, K={0..7,16..23}; lanes 16-31: row M=lane-16, K={8..15,24..31}
static __device__ __forceinline__ bf16x16 load_fragA(const unsigned short* p, int ld) {
  int lane = threadIdx.x & 31;
  int hf   = lane >> 4;
  int r    = lane & 15;
  const unsigned short* row = p + (size_t)r * ld + hf * 8;
  union { bf16x16 v; bf16x8 h[2]; } u;
  u.h[0] = *(const bf16x8*)(row);        // K = hf*8 .. hf*8+7
  u.h[1] = *(const bf16x8*)(row + 16);   // K = 16+hf*8 .. 16+hf*8+7
  return u.v;
}

// B fragment (32x16, KxN) loaded from B^T storage [N][K] with leading dim ld.
// lanes 0-15: col N=lane, K={0..7,16..23}; lanes 16-31: col N=lane-16, K={8..15,24..31}
static __device__ __forceinline__ bf16x16 load_fragB(const unsigned short* p, int ld) {
  int lane = threadIdx.x & 31;
  int hf   = lane >> 4;
  int r    = lane & 15;
  const unsigned short* row = p + (size_t)r * ld + hf * 8;
  union { bf16x16 v; bf16x8 h[2]; } u;
  u.h[0] = *(const bf16x8*)(row);        // K = hf*8 .. hf*8+7
  u.h[1] = *(const bf16x8*)(row + 16);   // K = 16+hf*8 .. 16+hf*8+7
  return u.v;
}

// ---------------------------------------------------------------------------
// 0) fp32 [K][N] -> bf16 transposed [N][K]
// ---------------------------------------------------------------------------
__global__ void convert_transpose_kernel(const float* __restrict__ W,
                                         unsigned short* __restrict__ Wt,
                                         int K, int N) {
  int idx = blockIdx.x * blockDim.x + threadIdx.x;
  if (idx >= K * N) return;
  int k = idx / N, n = idx % N;
  Wt[(size_t)n * K + k] = f2bf(W[idx]);
}

// ---------------------------------------------------------------------------
// 1) LayerNorm over E=1024, fp32 in -> bf16 out. One block per row.
// ---------------------------------------------------------------------------
__global__ void ln_kernel(const float* __restrict__ x,
                          const float* __restrict__ g,
                          const float* __restrict__ beta,
                          unsigned short* __restrict__ out) {
  int row = blockIdx.x;
  const float* xr = x + (size_t)row * E_;
  float vals[4], s = 0.f, ss = 0.f;
  for (int i = 0; i < 4; i++) {
    float v = xr[threadIdx.x + 256 * i];
    vals[i] = v; s += v; ss += v * v;
  }
  __shared__ float red0[256], red1[256];
  red0[threadIdx.x] = s; red1[threadIdx.x] = ss;
  __syncthreads();
  for (int off = 128; off > 0; off >>= 1) {
    if ((int)threadIdx.x < off) {
      red0[threadIdx.x] += red0[threadIdx.x + off];
      red1[threadIdx.x] += red1[threadIdx.x + off];
    }
    __syncthreads();
  }
  float mean = red0[0] * (1.f / E_);
  float var  = red1[0] * (1.f / E_) - mean * mean;
  float rstd = rsqrtf(var + 1e-5f);
  for (int i = 0; i < 4; i++) {
    int c = threadIdx.x + 256 * i;
    out[(size_t)row * E_ + c] = f2bf((vals[i] - mean) * rstd * g[c] + beta[c]);
  }
}

// ---------------------------------------------------------------------------
// 2) QKV projection: [B*S*H, 64] x [64,64] (shared per-head weights), WMMA.
//    Q,K stored [B,H,S,D]; V stored transposed [B,H,D,S].
// ---------------------------------------------------------------------------
__global__ void qkv_kernel(const unsigned short* __restrict__ nx,
                           const unsigned short* __restrict__ WqT,
                           const unsigned short* __restrict__ WkT,
                           const unsigned short* __restrict__ WvT,
                           unsigned short* __restrict__ Q,
                           unsigned short* __restrict__ Kout,
                           unsigned short* __restrict__ Vt) {
  int wave = blockIdx.x * (blockDim.x >> 5) + (threadIdx.x >> 5);
  int lane = threadIdx.x & 31;
  int hf = lane >> 4, l15 = lane & 15;
  int m0 = wave * 16;  // row in [B*S*H, 64]
  const unsigned short* Arow = nx + (size_t)m0 * D_;
  f32x8 aq[4] = {}, ak[4] = {}, av[4] = {};
  for (int k = 0; k < D_; k += 32) {
    bf16x16 a = load_fragA(Arow + k, D_);
    for (int nf = 0; nf < 4; nf++) {
      bf16x16 bq = load_fragB(WqT + (size_t)(nf * 16) * D_ + k, D_);
      bf16x16 bk = load_fragB(WkT + (size_t)(nf * 16) * D_ + k, D_);
      bf16x16 bv = load_fragB(WvT + (size_t)(nf * 16) * D_ + k, D_);
      aq[nf] = wmma_bf16(a, bq, aq[nf]);
      ak[nf] = wmma_bf16(a, bk, ak[nf]);
      av[nf] = wmma_bf16(a, bv, av[nf]);
    }
  }
  for (int nf = 0; nf < 4; nf++) {
    int d = nf * 16 + l15;
    for (int i = 0; i < 8; i++) {
      int m  = m0 + i + 8 * hf;
      int h  = m & (H_ - 1);
      int bs = m >> 4;              // /H_
      int b  = bs >> 10;            // /S_
      int s  = bs & (S_ - 1);
      size_t qk = ((size_t)(b * H_ + h) * S_ + s) * D_ + d;
      Q[qk]    = f2bf(aq[nf][i]);
      Kout[qk] = f2bf(ak[nf][i]);
      size_t vi = ((size_t)(b * H_ + h) * D_ + d) * S_ + s;
      Vt[vi] = f2bf(av[nf][i]);
    }
  }
}

// ---------------------------------------------------------------------------
// 3) Causal flash attention, scale = 1/sqrt(E) = 1/32.
//    One wave per (b,h,16-query tile); 32-key steps; P via LDS C->A relayout.
// ---------------------------------------------------------------------------
__global__ void attn_kernel(const unsigned short* __restrict__ Q,
                            const unsigned short* __restrict__ Kk,
                            const unsigned short* __restrict__ Vt,
                            unsigned short* __restrict__ O) {
  __shared__ unsigned short ldsP[8][16 * 32];   // per-wave 16x32 P tile (bf16)
  int wid  = threadIdx.x >> 5;
  int wave = blockIdx.x * 8 + wid;
  int lane = threadIdx.x & 31;
  int hf = lane >> 4, l15 = lane & 15;
  int qt = wave & (S_ / 16 - 1);
  int bh = wave >> 6;                            // b*H + h
  int qbase = qt * 16;

  const unsigned short* qp = Q + ((size_t)bh * S_ + qbase) * D_;
  bf16x16 qa0 = load_fragA(qp + 0,  D_);
  bf16x16 qa1 = load_fragA(qp + 32, D_);

  f32x8 oacc[4] = {};
  float rmax[8], rsum[8];
  for (int i = 0; i < 8; i++) { rmax[i] = -1e30f; rsum[i] = 0.f; }
  const float scale = 0.03125f;  // 1/sqrt(1024)

  int kend = qbase + 16;
  for (int k0 = 0; k0 < kend; k0 += 32) {
    f32x8 sc[2] = {};
    for (int sub = 0; sub < 2; sub++) {
      const unsigned short* kp = Kk + ((size_t)bh * S_ + (k0 + sub * 16)) * D_;
      sc[sub] = wmma_bf16(qa0, load_fragB(kp + 0,  D_), sc[sub]);
      sc[sub] = wmma_bf16(qa1, load_fragB(kp + 32, D_), sc[sub]);
    }
    float pv[2][8];
    for (int i = 0; i < 8; i++) {
      int qrow = qbase + i + 8 * hf;
      float l0 = sc[0][i] * scale;
      float l1 = sc[1][i] * scale;
      if (k0 + l15 > qrow)       l0 = -1e30f;
      if (k0 + 16 + l15 > qrow)  l1 = -1e30f;
      float cand = fmaxf(l0, l1);
      for (int m = 1; m < 16; m <<= 1) cand = fmaxf(cand, __shfl_xor(cand, m, 32));
      float nm   = fmaxf(rmax[i], cand);
      float corr = __expf(rmax[i] - nm);
      rmax[i] = nm;
      float p0 = __expf(l0 - nm);
      float p1 = __expf(l1 - nm);
      pv[0][i] = p0; pv[1][i] = p1;
      float ps = p0 + p1;
      for (int m = 1; m < 16; m <<= 1) ps += __shfl_xor(ps, m, 32);
      rsum[i] = rsum[i] * corr + ps;
      for (int nf = 0; nf < 4; nf++) oacc[nf][i] *= corr;
    }
    for (int sub = 0; sub < 2; sub++)
      for (int i = 0; i < 8; i++)
        ldsP[wid][(i + 8 * hf) * 32 + sub * 16 + l15] = f2bf(pv[sub][i]);
    asm volatile("s_wait_dscnt 0" ::: "memory");
    bf16x16 pA = load_fragA(&ldsP[wid][0], 32);
    for (int nf = 0; nf < 4; nf++) {
      bf16x16 vb = load_fragB(Vt + ((size_t)bh * D_ + nf * 16) * S_ + k0, S_);
      oacc[nf] = wmma_bf16(pA, vb, oacc[nf]);
    }
    asm volatile("s_wait_dscnt 0" ::: "memory");
  }

  int b = bh >> 4, h = bh & (H_ - 1);
  float inv[8];
  for (int i = 0; i < 8; i++) inv[i] = 1.f / rsum[i];
  for (int nf = 0; nf < 4; nf++) {
    int d = nf * 16 + l15;
    for (int i = 0; i < 8; i++) {
      int srow = qbase + i + 8 * hf;
      O[((size_t)(b * S_ + srow)) * E_ + h * D_ + d] = f2bf(oacc[nf][i] * inv[i]);
    }
  }
}

// ---------------------------------------------------------------------------
// 4) GEMM  C = A(bf16 [M][K]) * B(bf16 B^T [N][K]) + bias (+epilogue)
//    Block (256 thr, 8 waves as 4M x 2N) computes a 256x128 macro-tile.
//    B panel (128 n-rows x 32 k) staged global->LDS with async loads
//    (global_load_async_to_lds_b128, ASYNCcnt), DOUBLE-BUFFERED so panel s+1
//    streams in behind the 16 WMMAs of panel s. One barrier per k-step.
//    MODE 0: out fp32 = v + resid ; MODE 1: out bf16 = relu(v)
// ---------------------------------------------------------------------------
template <int MODE>
__global__ void gemm_kernel(const unsigned short* __restrict__ A,
                            const unsigned short* __restrict__ Bt,
                            const float* __restrict__ bias,
                            const float* __restrict__ resid,
                            float* __restrict__ outF,
                            unsigned short* __restrict__ outH,
                            int N, int K, int tiles_n) {
  __shared__ unsigned short ldsB[2][128 * 32];   // 2 x 8 KB B^T panels
  int wid  = threadIdx.x >> 5;
  int lane = threadIdx.x & 31;
  int hf = lane >> 4, l15 = lane & 15;
  int wm = wid >> 1, wn = wid & 1;
  int tm = blockIdx.x / tiles_n, tn = blockIdx.x % tiles_n;
  int m0 = tm * 256 + wm * 64;
  int n0 = tn * 128;                 // block-level N base (LDS panel origin)
  int nw = n0 + wn * 64;             // wave-level N base

  // Stage panel for k-step `s` into buffer `buf`: 512 x 16B chunks, 2/thread.
  auto stage = [&](int s, int buf) {
    for (int c = threadIdx.x; c < 512; c += 256) {
      int row = c >> 2, qtr = c & 3;
      unsigned ldst = (unsigned)(size_t)(&ldsB[buf][row * 32 + qtr * 8]);
      unsigned long long ga =
          (unsigned long long)(size_t)(Bt + (size_t)(n0 + row) * K + s * 32 + qtr * 8);
      asm volatile("global_load_async_to_lds_b128 %0, %1, off"
                   :: "v"(ldst), "v"(ga) : "memory");
    }
  };

  int nsteps = K / 32;
  stage(0, 0);                        // prologue: panel 0 in flight
  f32x8 acc[4][4] = {};
  for (int s = 0; s < nsteps; s++) {
    int buf = s & 1;
    int k = s * 32;
    // wait for this wave's panel-s loads, then make them block-visible
    asm volatile("s_wait_asynccnt 0x0" ::: "memory");
    __syncthreads();
    // kick off panel s+1 into the other buffer; overlaps with compute below
    if (s + 1 < nsteps) stage(s + 1, buf ^ 1);
    // prefetch A stream two k-steps ahead (speculative, global_prefetch_b8)
    if (k + 64 < K) __builtin_prefetch(A + (size_t)m0 * K + k + 64, 0, 0);

    bf16x16 a0 = load_fragA(A + (size_t)(m0 +  0) * K + k, K);
    bf16x16 a1 = load_fragA(A + (size_t)(m0 + 16) * K + k, K);
    bf16x16 a2 = load_fragA(A + (size_t)(m0 + 32) * K + k, K);
    bf16x16 a3 = load_fragA(A + (size_t)(m0 + 48) * K + k, K);
    for (int nf = 0; nf < 4; nf++) {
      bf16x16 b = load_fragB(&ldsB[buf][(wn * 64 + nf * 16) * 32], 32);
      acc[0][nf] = wmma_bf16(a0, b, acc[0][nf]);
      acc[1][nf] = wmma_bf16(a1, b, acc[1][nf]);
      acc[2][nf] = wmma_bf16(a2, b, acc[2][nf]);
      acc[3][nf] = wmma_bf16(a3, b, acc[3][nf]);
    }
    // (next iteration's barrier orders these ds reads vs. panel overwrite:
    //  all reads are dscnt-waited before their consuming WMMAs above)
  }

  for (int mi = 0; mi < 4; mi++) {
    for (int nf = 0; nf < 4; nf++) {
      int n = nw + nf * 16 + l15;
      float bi = bias[n];
      for (int i = 0; i < 8; i++) {
        int m = m0 + mi * 16 + i + 8 * hf;
        size_t idx = (size_t)m * N + n;
        float v = acc[mi][nf][i] + bi;
        if (MODE == 0) outF[idx] = v + resid[idx];
        else           outH[idx] = f2bf(fmaxf(v, 0.f));
      }
    }
  }
}

// ---------------------------------------------------------------------------
// Host side
// ---------------------------------------------------------------------------
extern "C" void kernel_launch(void* const* d_in, const int* in_sizes, int n_in,
                              void* d_out, int out_size, void* d_ws, size_t ws_size,
                              hipStream_t stream) {
  const float* x    = (const float*)d_in[0];
  const float* Wq   = (const float*)d_in[1];
  const float* Wk   = (const float*)d_in[2];
  const float* Wv   = (const float*)d_in[3];
  const float* Wo   = (const float*)d_in[4];
  const float* bo   = (const float*)d_in[5];
  const float* W1   = (const float*)d_in[6];
  const float* b1   = (const float*)d_in[7];
  const float* W2   = (const float*)d_in[8];
  const float* b2   = (const float*)d_in[9];
  const float* g1   = (const float*)d_in[10];
  const float* be1  = (const float*)d_in[11];
  const float* g2   = (const float*)d_in[12];
  const float* be2  = (const float*)d_in[13];
  float* out = (float*)d_out;

  size_t off = 0;
  auto alloc = [&](size_t bytes) -> void* {
    void* p = (char*)d_ws + off;
    off += (bytes + 255) & ~(size_t)255;
    return p;
  };
  const size_t BSE = (size_t)B_ * S_ * E_;
  unsigned short* nx  = (unsigned short*)alloc(BSE * 2);           // LN output (reused for LN2)
  unsigned short* qb  = (unsigned short*)alloc(BSE * 2);           // Q [B,H,S,D]
  unsigned short* kb  = (unsigned short*)alloc(BSE * 2);           // K [B,H,S,D]
  unsigned short* vtb = (unsigned short*)alloc(BSE * 2);           // V^T [B,H,D,S]
  unsigned short* ob  = (unsigned short*)alloc(BSE * 2);           // attn out [B*S][E]
  float*          x1  = (float*)alloc(BSE * 4);                    // post-attn residual
  unsigned short* hb  = (unsigned short*)alloc((size_t)B_ * S_ * F_ * 2); // MLP hidden
  unsigned short* WqT = (unsigned short*)alloc((size_t)D_ * D_ * 2);
  unsigned short* WkT = (unsigned short*)alloc((size_t)D_ * D_ * 2);
  unsigned short* WvT = (unsigned short*)alloc((size_t)D_ * D_ * 2);
  unsigned short* WoT = (unsigned short*)alloc((size_t)E_ * E_ * 2);
  unsigned short* W1T = (unsigned short*)alloc((size_t)E_ * F_ * 2);
  unsigned short* W2T = (unsigned short*)alloc((size_t)F_ * E_ * 2);

  // Weight conversion + transpose (bf16, [N][K])
  convert_transpose_kernel<<<(D_ * D_ + 255) / 256, 256, 0, stream>>>(Wq, WqT, D_, D_);
  convert_transpose_kernel<<<(D_ * D_ + 255) / 256, 256, 0, stream>>>(Wk, WkT, D_, D_);
  convert_transpose_kernel<<<(D_ * D_ + 255) / 256, 256, 0, stream>>>(Wv, WvT, D_, D_);
  convert_transpose_kernel<<<(E_ * E_ + 255) / 256, 256, 0, stream>>>(Wo, WoT, E_, E_);
  convert_transpose_kernel<<<(E_ * F_ + 255) / 256, 256, 0, stream>>>(W1, W1T, E_, F_);
  convert_transpose_kernel<<<(F_ * E_ + 255) / 256, 256, 0, stream>>>(W2, W2T, F_, E_);

  const int rows = B_ * S_;  // 8192
  // LN1
  ln_kernel<<<rows, 256, 0, stream>>>(x, g1, be1, nx);
  // QKV: waves = B*S*H/16 = 8192 -> 1024 blocks of 8 waves
  qkv_kernel<<<1024, 256, 0, stream>>>(nx, WqT, WkT, WvT, qb, kb, vtb);
  // Flash attention: waves = B*H*(S/16) = 8192 -> 1024 blocks
  attn_kernel<<<1024, 256, 0, stream>>>(qb, kb, vtb, ob);
  // x1 = x + O @ Wo + bo : macro tiles = (8192/256)*(1024/128) = 32*8 = 256 blocks
  gemm_kernel<0><<<256, 256, 0, stream>>>(ob, WoT, bo, x, x1, nullptr, E_, E_, E_ / 128);
  // LN2 (reuse nx)
  ln_kernel<<<rows, 256, 0, stream>>>(x1, g2, be2, nx);
  // h = relu(nx @ W1 + b1) : 32*32 = 1024 blocks
  gemm_kernel<1><<<1024, 256, 0, stream>>>(nx, W1T, b1, nullptr, nullptr, hb, F_, E_, F_ / 128);
  // out = x1 + h @ W2 + b2 : 32*8 = 256 blocks
  gemm_kernel<0><<<256, 256, 0, stream>>>(hb, W2T, b2, x1, out, nullptr, E_, F_, E_ / 128);
}
